// SelfMultiheadAttention_12378095747770
// MI455X (gfx1250) — matrix operand order repaired and checked by reading, not compiled, not run on hardware
//
#include <hip/hip_runtime.h>

// Problem constants (match reference)
#define SEQ    2048
#define BATCH  2
#define EMS    256
#define NHEAD  8
#define HEAD   32
#define TOKENS (SEQ * BATCH)
#define NCHUNK (SEQ / 32)

typedef __bf16 v16bf __attribute__((ext_vector_type(16)));
typedef __bf16 bf8   __attribute__((ext_vector_type(8)));
typedef float  v8f   __attribute__((ext_vector_type(8)));
typedef float  f4    __attribute__((ext_vector_type(4)));
typedef int    v4i   __attribute__((ext_vector_type(4)));

// ---- gfx1250 async global->LDS copy (ASYNCcnt-tracked), with sync fallback ----
#if __has_builtin(__builtin_amdgcn_global_load_async_to_lds_b128)
#define HAVE_ASYNC_LDS 1
__device__ __forceinline__ void cp16_async(const void* g, void* l) {
    __builtin_amdgcn_global_load_async_to_lds_b128((v4i*)g, (v4i*)l, 0, 0);
}
__device__ __forceinline__ void wait_async() {
    asm volatile("s_wait_asynccnt 0x0" ::: "memory");
}
#else
#define HAVE_ASYNC_LDS 0
__device__ __forceinline__ void cp16_async(const void* g, void* l) {
    *(f4*)l = *(const f4*)g;            // synchronous fallback
}
__device__ __forceinline__ void wait_async() {}
#endif

// ---------------------------------------------------------------------------
// QKV projection: D[16 tokens x 16 features] = X * W^T + b, bf16 WMMA, f32 acc
// X flat token index t = s*BATCH + b (natural em [S,B,E] layout).
// Output: q/k -> [B][H][S][D] bf16 ; v -> transposed [B][H][D][S] bf16.
// One wave (32 threads) per 16x16 output tile.
// ---------------------------------------------------------------------------
__global__ __launch_bounds__(32)
void qkv_proj_kernel(const float* __restrict__ X,
                     const float* __restrict__ W,
                     const float* __restrict__ bias,
                     __bf16* __restrict__ dst,
                     int vTranspose)
{
    const int lane = threadIdx.x;
    const int grp  = lane >> 4;   // 0: lanes 0-15, 1: lanes 16-31
    const int l16  = lane & 15;
    const int m0   = blockIdx.x * 16;   // token tile
    const int n0   = blockIdx.y * 16;   // output-feature tile

    v8f acc = {};
    #pragma unroll
    for (int k0 = 0; k0 < EMS; k0 += 32) {
        // A fragment (16x32 bf16): lane row = l16;
        // element j -> K = k0 + grp*16 + j
        const float* arow = X + (size_t)(m0 + l16) * EMS + k0 + grp * 16;
        v16bf a;
        #pragma unroll
        for (int j = 0; j < 8; ++j) {
            a[j]     = (__bf16)arow[j];
            a[j + 8] = (__bf16)arow[8 + j];
        }
        // B fragment (32x16 bf16): lane col n = n0+l16;
        // element j -> K = k0 + grp*16 + j ; B[k][n] = W[n][k]
        const float* brow = W + (size_t)(n0 + l16) * EMS + k0 + grp * 16;
        v16bf bfr;
        #pragma unroll
        for (int j = 0; j < 16; ++j) bfr[j] = (__bf16)brow[j];

        acc = __builtin_amdgcn_wmma_f32_16x16x32_bf16(
                  false, a, false, bfr, (short)0, acc, false, false);
    }

    const int   n  = n0 + l16;
    const int   h  = n / HEAD;
    const int   d  = n % HEAD;
    const float bn = bias[n];

    #pragma unroll
    for (int i = 0; i < 8; ++i) {
        const int t = m0 + i + 8 * grp;         // D row = token
        const int s = t / BATCH;
        const int b = t % BATCH;
        const __bf16 val = (__bf16)(acc[i] + bn);
        if (vTranspose)
            dst[(((size_t)b * NHEAD + h) * HEAD + d) * SEQ + s] = val;
        else
            dst[(((size_t)b * NHEAD + h) * SEQ + s) * HEAD + d] = val;
    }
}

// ---------------------------------------------------------------------------
// Flash attention: block = 8 waves = 8 heads over the same (b, 16-row q-tile).
// Mask tile [16 x 32] f32 is staged in LDS once per chunk (async prefetch,
// double-buffered, issued by wave 0) and shared by all heads.
// Per chunk & wave: 2 WMMAs for scores (K-dim = D = 32), online softmax in
// f32, P transposed through per-wave LDS, 2 accumulating WMMAs against Vt.
// ---------------------------------------------------------------------------
__global__ __launch_bounds__(256)
void fa_kernel(const __bf16* __restrict__ Q,    // [B][H][S][D]
               const __bf16* __restrict__ Km,   // [B][H][S][D]
               const __bf16* __restrict__ Vt,   // [B][H][D][S]
               const float*  __restrict__ mask, // [B][1][S][S]
               float* __restrict__ out)         // [S][B][E]
{
    __shared__ float  ldsM[2][16 * 32];          // mask tile double buffer (4 KB)
    __shared__ __bf16 ldsP[NHEAD][16 * 32];      // per-wave P staging (8 KB)

    const int tid  = threadIdx.x;
    const int wave = tid >> 5;                   // = head index
    const int lane = tid & 31;
    const int grp  = lane >> 4;
    const int l16  = lane & 15;
    const int q0   = blockIdx.x * 16;
    const int b    = blockIdx.y;
    const int h    = wave;
    const int bh   = b * NHEAD + h;

    const __bf16* Qbh = Q  + (size_t)bh * SEQ * HEAD;
    const __bf16* Kbh = Km + (size_t)bh * SEQ * HEAD;
    const __bf16* Vbh = Vt + (size_t)bh * HEAD * SEQ;
    const float*  Mb  = mask + (size_t)b * SEQ * SEQ;

    // Q A-fragment (16 rows x D=32), loaded once, reused all chunks.
    v16bf qa;
    {
        const __bf16* qrow = Qbh + (size_t)(q0 + l16) * HEAD + grp * 16;
        bf8 lo = *(const bf8*)(qrow);
        bf8 hi = *(const bf8*)(qrow + 8);
        #pragma unroll
        for (int j = 0; j < 8; ++j) { qa[j] = lo[j]; qa[j + 8] = hi[j]; }
    }

    // Wave 0: prefetch first mask tile (2048 B = 4 x 32 lanes x 16 B).
    if (wave == 0) {
        #pragma unroll
        for (int k = 0; k < 4; ++k) {
            const int tb  = (k * 32 + lane) * 16;     // byte index in tile
            const int row = tb >> 7;                  // /128 B per row
            const int col = tb & 127;
            const char* g = (const char*)(Mb + (size_t)(q0 + row) * SEQ) + col;
            cp16_async(g, (char*)&ldsM[0][0] + tb);
        }
        wait_async();
    }
    __syncthreads();

    v8f ctx0 = {}, ctx1 = {};
    float m_run[8], l_run[8];
    #pragma unroll
    for (int i = 0; i < 8; ++i) { m_run[i] = -1e30f; l_run[i] = 0.0f; }

    const float scale = 0.17677669529663687f;   // 1/sqrt(32)

    for (int c = 0; c < NCHUNK; ++c) {
        const int kv0 = c * 32;
        const int buf = c & 1;

        // Wave 0: async-prefetch next chunk's mask tile into the other buffer.
        if (wave == 0 && (c + 1) < NCHUNK) {
            #pragma unroll
            for (int k = 0; k < 4; ++k) {
                const int tb  = (k * 32 + lane) * 16;
                const int row = tb >> 7;
                const int col = tb & 127;
                const char* g = (const char*)(Mb + (size_t)(q0 + row) * SEQ + kv0 + 32) + col;
                cp16_async(g, (char*)&ldsM[buf ^ 1][0] + tb);
            }
        }

        // K B-fragments: tile0 = kv cols [kv0,kv0+16), tile1 = [kv0+16,kv0+32)
        v16bf kb0, kb1;
        {
            const __bf16* kr0 = Kbh + (size_t)(kv0 + l16) * HEAD + grp * 16;
            const __bf16* kr1 = Kbh + (size_t)(kv0 + 16 + l16) * HEAD + grp * 16;
            bf8 a0 = *(const bf8*)kr0, c0 = *(const bf8*)(kr0 + 8);
            bf8 a1 = *(const bf8*)kr1, c1 = *(const bf8*)(kr1 + 8);
            #pragma unroll
            for (int j = 0; j < 8; ++j) {
                kb0[j] = a0[j]; kb0[j + 8] = c0[j];
                kb1[j] = a1[j]; kb1[j + 8] = c1[j];
            }
        }

        v8f z = {};
        v8f s0 = __builtin_amdgcn_wmma_f32_16x16x32_bf16(
                     false, qa, false, kb0, (short)0, z, false, false);
        v8f s1 = __builtin_amdgcn_wmma_f32_16x16x32_bf16(
                     false, qa, false, kb1, (short)0, z, false, false);

        // Online softmax update, per D-fragment row (row = i + 8*grp).
        #pragma unroll
        for (int i = 0; i < 8; ++i) {
            const int rl = i + 8 * grp;             // local row in tile
            float v0 = s0[i] * scale + ldsM[buf][rl * 32 + l16];
            float v1 = s1[i] * scale + ldsM[buf][rl * 32 + 16 + l16];

            float mx = fmaxf(v0, v1);
            mx = fmaxf(mx, __shfl_xor(mx, 1));
            mx = fmaxf(mx, __shfl_xor(mx, 2));
            mx = fmaxf(mx, __shfl_xor(mx, 4));
            mx = fmaxf(mx, __shfl_xor(mx, 8));
            const float mnew = fmaxf(m_run[i], mx);
            const float corr = __expf(m_run[i] - mnew);

            const float p0 = __expf(v0 - mnew);
            const float p1 = __expf(v1 - mnew);
            float rs = p0 + p1;
            rs += __shfl_xor(rs, 1);
            rs += __shfl_xor(rs, 2);
            rs += __shfl_xor(rs, 4);
            rs += __shfl_xor(rs, 8);

            l_run[i] = l_run[i] * corr + rs;
            m_run[i] = mnew;
            ctx0[i] *= corr;
            ctx1[i] *= corr;

            // Stage P (bf16) to per-wave LDS [16][32] for A-frag reload.
            ldsP[wave][rl * 32 + l16]      = (__bf16)p0;
            ldsP[wave][rl * 32 + 16 + l16] = (__bf16)p1;
        }
        __syncthreads();   // order cross-lane P exchange (and align waves)

        // P A-fragment (16 q-rows x 32 kv): lane row = l16;
        // element j -> kv = grp*16 + j
        v16bf pa;
        {
            const __bf16* prow = &ldsP[wave][0] + l16 * 32 + grp * 16;
            bf8 lo = *(const bf8*)(prow);
            bf8 hi = *(const bf8*)(prow + 8);
            #pragma unroll
            for (int j = 0; j < 8; ++j) { pa[j] = lo[j]; pa[j + 8] = hi[j]; }
        }

        // V B-fragments from transposed V: ctx0 covers d=l16, ctx1 d=16+l16;
        // element j -> kv = kv0 + grp*16 + j (contiguous in Vt rows).
        v16bf vb0, vb1;
        {
            const __bf16* vr0 = Vbh + (size_t)(l16) * SEQ + kv0 + grp * 16;
            const __bf16* vr1 = Vbh + (size_t)(16 + l16) * SEQ + kv0 + grp * 16;
            bf8 a0 = *(const bf8*)vr0, c0 = *(const bf8*)(vr0 + 8);
            bf8 a1 = *(const bf8*)vr1, c1 = *(const bf8*)(vr1 + 8);
            #pragma unroll
            for (int j = 0; j < 8; ++j) {
                vb0[j] = a0[j]; vb0[j + 8] = c0[j];
                vb1[j] = a1[j]; vb1[j + 8] = c1[j];
            }
        }

        ctx0 = __builtin_amdgcn_wmma_f32_16x16x32_bf16(
                   false, pa, false, vb0, (short)0, ctx0, false, false);
        ctx1 = __builtin_amdgcn_wmma_f32_16x16x32_bf16(
                   false, pa, false, vb1, (short)0, ctx1, false, false);

        if (wave == 0) wait_async();   // next mask tile landed in LDS
        __syncthreads();               // publish it to all waves
    }

    // Normalize and write out[s][b][h*32 + d], fp32.
    #pragma unroll
    for (int i = 0; i < 8; ++i) {
        const int   s   = q0 + i + 8 * grp;
        const float inv = 1.0f / l_run[i];
        float* orow = out + (size_t)s * BATCH * EMS + (size_t)b * EMS + h * HEAD;
        orow[l16]      = ctx0[i] * inv;
        orow[16 + l16] = ctx1[i] * inv;
    }
}

// ---------------------------------------------------------------------------
// Host launcher
// ---------------------------------------------------------------------------
extern "C" void kernel_launch(void* const* d_in, const int* in_sizes, int n_in,
                              void* d_out, int out_size, void* d_ws, size_t ws_size,
                              hipStream_t stream)
{
    (void)in_sizes; (void)n_in; (void)out_size; (void)ws_size;

    const float* em   = (const float*)d_in[0];
    const float* mask = (const float*)d_in[1];
    const float* Wq   = (const float*)d_in[2];
    const float* bq   = (const float*)d_in[3];
    const float* Wk   = (const float*)d_in[4];
    const float* bk   = (const float*)d_in[5];
    const float* Wv   = (const float*)d_in[6];
    const float* bv   = (const float*)d_in[7];
    float* out = (float*)d_out;

    const size_t perMat = (size_t)BATCH * NHEAD * SEQ * HEAD;  // 2 MB each (bf16)
    __bf16* Qw = (__bf16*)d_ws;
    __bf16* Kw = Qw + perMat;
    __bf16* Vw = Kw + perMat;

    dim3 gp(TOKENS / 16, EMS / 16);
    qkv_proj_kernel<<<gp, 32, 0, stream>>>(em, Wq, bq, Qw, 0);
    qkv_proj_kernel<<<gp, 32, 0, stream>>>(em, Wk, bk, Kw, 0);
    qkv_proj_kernel<<<gp, 32, 0, stream>>>(em, Wv, bv, Vw, 1);

    fa_kernel<<<dim3(SEQ / 16, BATCH), 256, 0, stream>>>(Qw, Kw, Vw, mask, out);
}
